// HeteroGNN_50981261804006
// MI455X (gfx1250) — matrix operand order repaired and checked by reading, not compiled, not run on hardware
//
#include <hip/hip_runtime.h>
#include <math.h>

typedef _Float16 h8   __attribute__((ext_vector_type(8)));
typedef _Float16 v16h __attribute__((ext_vector_type(16)));
typedef float    v8f  __attribute__((ext_vector_type(8)));

__device__ __forceinline__ float gelu_erf(float x) {
    return 0.5f * x * (1.0f + erff(x * 0.7071067811865475f));
}

// ---------------------------------------------------------------------------
// WMMA GEMM: C[M,NCOLS] = A[M,K] @ W[K,NCOLS]  (f32 in/out, f16 WMMA compute)
// Block: 256 threads (8 waves), 32 output rows per block.
// A staged in LDS row-major [32][K+8]; W staged transposed [NCOLS][K+8] so
// both fragments load as two aligned 16-byte chunks per lane, matching the
// CDNA5 16-bit A (16x32) / B (32x16) VGPR layouts.
// ---------------------------------------------------------------------------
template<int K, int NCOLS, bool BIAS_GELU>
__global__ __launch_bounds__(256) void gemm_wmma(const float* __restrict__ A,
                                                 const float* __restrict__ W,
                                                 const float* __restrict__ bias,
                                                 float* __restrict__ C, int M)
{
    constexpr int LDA = K + 8;  // pad 16B to dodge bank conflicts, keep 16B align
    __shared__ alignas(16) _Float16 sA[32 * LDA];
    __shared__ alignas(16) _Float16 sW[NCOLS * LDA];  // transposed: [col][k]

    const int tid  = threadIdx.x;
    const int row0 = blockIdx.x * 32;

    for (int idx = tid; idx < 32 * K; idx += 256) {
        int r = idx / K, c = idx % K;
        int gr = row0 + r;
        float v = (gr < M) ? A[(size_t)gr * K + c] : 0.0f;
        sA[r * LDA + c] = (_Float16)v;
    }
    for (int idx = tid; idx < K * NCOLS; idx += 256) {
        int k = idx / NCOLS, c = idx % NCOLS;
        sW[c * LDA + k] = (_Float16)W[idx];
    }
    __syncthreads();

    const int wave  = tid >> 5;
    const int lane  = tid & 31;
    const int lhalf = (lane < 16) ? 0 : 1;   // which K half-group this lane holds
    const int l16   = lane & 15;
    constexpr int NT = NCOLS / 16;           // col tiles; 2 row tiles of 16

    for (int t = wave; t < 2 * NT; t += 8) {
        const int rt = t / NT;
        const int ct = t % NT;
        v8f acc = {};
        for (int k0 = 0; k0 < K; k0 += 32) {
            const int arow = rt * 16 + l16;
            // A 16x32 f16 layout: lanes<16 hold K {0..7,16..23}, lanes>=16 {8..15,24..31}
            h8 alo = *(const h8*)&sA[arow * LDA + k0 + lhalf * 8];
            h8 ahi = *(const h8*)&sA[arow * LDA + k0 + 16 + lhalf * 8];
            const int bcol = ct * 16 + l16;
            // B 32x16 f16 layout: lanes<16 hold K {0..7,16..23}, lanes>=16 {8..15,24..31}
            h8 blo = *(const h8*)&sW[bcol * LDA + k0 + lhalf * 8];
            h8 bhi = *(const h8*)&sW[bcol * LDA + k0 + 16 + lhalf * 8];
            v16h a, b;
            #pragma unroll
            for (int i = 0; i < 8; ++i) {
                a[i] = alo[i]; a[8 + i] = ahi[i];
                b[i] = blo[i]; b[8 + i] = bhi[i];
            }
            acc = __builtin_amdgcn_wmma_f32_16x16x32_f16(
                false, a, false, b, (short)0, acc, false, false);
        }
        const int colg = ct * 16 + l16;
        #pragma unroll
        for (int i = 0; i < 8; ++i) {
            int rowg = row0 + rt * 16 + lhalf * 8 + i;
            if (rowg < M) {
                float v = acc[i];
                if constexpr (BIAS_GELU) v = gelu_erf(v + bias[colg]);
                C[(size_t)rowg * NCOLS + colg] = v;
            }
        }
    }
}

// ---------------------------------------------------------------------------
// a_src[n,h] = <h1[n,h,:], att_src[h,:]> ; same for dst.  One wave per (n,h).
// ---------------------------------------------------------------------------
template<int H>
__global__ void att_coef(const float* __restrict__ h, const float* __restrict__ att_s,
                         const float* __restrict__ att_d, float* __restrict__ a_s,
                         float* __restrict__ a_d, int N)
{
    int gw   = (int)((blockIdx.x * (unsigned)blockDim.x + threadIdx.x) >> 5);
    int lane = threadIdx.x & 31;
    if (gw >= N * H) return;
    int n = gw / H, hd = gw % H;
    const float* hp = h + ((size_t)n * H + hd) * 64;
    float s = hp[lane] * att_s[hd * 64 + lane] + hp[lane + 32] * att_s[hd * 64 + 32 + lane];
    float d = hp[lane] * att_d[hd * 64 + lane] + hp[lane + 32] * att_d[hd * 64 + 32 + lane];
    #pragma unroll
    for (int off = 16; off > 0; off >>= 1) {
        s += __shfl_down(s, off, 32);
        d += __shfl_down(d, off, 32);
    }
    if (lane == 0) { a_s[gw] = s; a_d[gw] = d; }
}

// order-preserving float<->uint map for atomicMax-based segment max
__device__ __forceinline__ unsigned fkey(float f) {
    unsigned u = __float_as_uint(f);
    return (u & 0x80000000u) ? ~u : (u | 0x80000000u);
}
__device__ __forceinline__ float finv(unsigned u) {
    return (u & 0x80000000u) ? __uint_as_float(u & 0x7fffffffu) : __uint_as_float(~u);
}

template<int H>
__global__ void edge_logit(const int* __restrict__ src, const int* __restrict__ dst,
                           int E, int Etot,
                           const float* __restrict__ a_s, const float* __restrict__ a_d,
                           float* __restrict__ logits, unsigned* __restrict__ segmax)
{
    int e = blockIdx.x * blockDim.x + threadIdx.x;
    if (e >= Etot) return;
    int s = (e < E) ? src[e] : (e - E);
    int d = (e < E) ? dst[e] : (e - E);
    #pragma unroll
    for (int hd = 0; hd < H; ++hd) {
        float l = a_s[s * H + hd] + a_d[d * H + hd];
        l = (l > 0.0f) ? l : 0.2f * l;            // leaky_relu(0.2)
        logits[(size_t)e * H + hd] = l;
        atomicMax(&segmax[d * H + hd], fkey(l));
    }
}

template<int H>
__global__ void edge_exp(const int* __restrict__ src, const int* __restrict__ dst,
                         int E, int Etot,
                         const unsigned* __restrict__ segmax, float* __restrict__ logits,
                         float* __restrict__ denom)
{
    int e = blockIdx.x * blockDim.x + threadIdx.x;
    if (e >= Etot) return;
    int d = (e < E) ? dst[e] : (e - E);
    #pragma unroll
    for (int hd = 0; hd < H; ++hd) {
        float l  = logits[(size_t)e * H + hd];
        float m  = finv(segmax[d * H + hd]);
        float ex = expf(l - m);
        logits[(size_t)e * H + hd] = ex;
        atomicAdd(&denom[d * H + hd], ex);
    }
}

// one thread per (edge, head, channel): coalesced gather + coalesced atomics
template<int H>
__global__ void edge_msg(const int* __restrict__ src, const int* __restrict__ dst,
                         int E, int Etot,
                         const float* __restrict__ h, const float* __restrict__ ealpha,
                         const float* __restrict__ denom, float* __restrict__ agg)
{
    long long gid = (long long)blockIdx.x * blockDim.x + threadIdx.x;
    long long total = (long long)Etot * H * 64;
    if (gid >= total) return;
    int c  = (int)(gid & 63);
    int hd = (int)((gid >> 6) % H);
    long long e = gid / (64LL * H);
    int s = (e < E) ? src[e] : (int)(e - E);
    int d = (e < E) ? dst[e] : (int)(e - E);
    float alpha = ealpha[(size_t)e * H + hd] / (denom[d * H + hd] + 1e-16f);
    float v = h[((size_t)s * H + hd) * 64 + c] * alpha;
    atomicAdd(&agg[((size_t)d * H + hd) * 64 + c], v);
}

// h(in/out): h <- gelu(mean_heads(agg) + b1 + h)
__global__ void res1(const float* __restrict__ agg, const float* __restrict__ b1,
                     float* __restrict__ h, int N)
{
    int i = blockIdx.x * blockDim.x + threadIdx.x;
    if (i >= N * 64) return;
    int c = i & 63, n = i >> 6;
    float m = 0.5f * (agg[(size_t)n * 128 + c] + agg[(size_t)n * 128 + 64 + c]);
    h[i] = gelu_erf(m + b1[c] + h[i]);
}

__global__ void bias_add(float* __restrict__ out, const float* __restrict__ b, int N)
{
    int i = blockIdx.x * blockDim.x + threadIdx.x;
    if (i >= N * 64) return;
    out[i] += b[i & 63];
}

extern "C" void kernel_launch(void* const* d_in, const int* in_sizes, int n_in,
                              void* d_out, int out_size, void* d_ws, size_t ws_size,
                              hipStream_t stream)
{
    const float* x        = (const float*)d_in[0];
    const int*   ei       = (const int*)d_in[1];
    const float* lin_w    = (const float*)d_in[2];
    const float* lin_b    = (const float*)d_in[3];
    const float* w1       = (const float*)d_in[4];
    const float* att_src1 = (const float*)d_in[5];
    const float* att_dst1 = (const float*)d_in[6];
    const float* b1       = (const float*)d_in[7];
    const float* w2       = (const float*)d_in[8];
    const float* att_src2 = (const float*)d_in[9];
    const float* att_dst2 = (const float*)d_in[10];
    const float* b2       = (const float*)d_in[11];
    float* out = (float*)d_out;

    const int N    = in_sizes[0] / 128;
    const int E    = in_sizes[1] / 2;
    const int Etot = E + N;
    const int* src = ei;
    const int* dst = ei + E;

    // workspace layout (floats)
    float*    ws   = (float*)d_ws;
    float*    h0   = ws;                               // [N,64]  (becomes h2 in place)
    float*    h1   = h0   + (size_t)N * 64;            // [N,128] (reused as h3 [N,64])
    float*    agg1 = h1   + (size_t)N * 128;           // [N,2,64]
    float*    a1s  = agg1 + (size_t)N * 128;           // [N,2] (layer2: [N,1])
    float*    a1d  = a1s  + (size_t)N * 2;             // [N,2]
    unsigned* m1   = (unsigned*)(a1d + (size_t)N * 2); // [N,2] keys
    float*    dn1  = (float*)(m1 + (size_t)N * 2);     // [N,2]
    float*    e1   = dn1  + (size_t)N * 2;             // [Etot,2] (layer2: [Etot,1])

    const int gemm_blocks = (N + 31) / 32;
    const int eb          = (Etot + 255) / 256;

    // ---- input projection: h0 = gelu(x @ lin_w + lin_b) ----
    gemm_wmma<128, 64, true><<<gemm_blocks, 256, 0, stream>>>(x, lin_w, lin_b, h0, N);

    // ---- conv1 (heads=2) ----
    gemm_wmma<64, 128, false><<<gemm_blocks, 256, 0, stream>>>(h0, w1, nullptr, h1, N);
    {
        int waves  = N * 2;
        int blocks = (waves * 32 + 255) / 256;
        att_coef<2><<<blocks, 256, 0, stream>>>(h1, att_src1, att_dst1, a1s, a1d, N);
    }
    hipMemsetAsync(m1,   0, (size_t)N * 2 * sizeof(unsigned), stream);
    hipMemsetAsync(dn1,  0, (size_t)N * 2 * sizeof(float),    stream);
    hipMemsetAsync(agg1, 0, (size_t)N * 128 * sizeof(float),  stream);
    edge_logit<2><<<eb, 256, 0, stream>>>(src, dst, E, Etot, a1s, a1d, e1, m1);
    edge_exp<2><<<eb, 256, 0, stream>>>(src, dst, E, Etot, m1, e1, dn1);
    {
        long long total  = (long long)Etot * 2 * 64;
        int       blocks = (int)((total + 255) / 256);
        edge_msg<2><<<blocks, 256, 0, stream>>>(src, dst, E, Etot, h1, e1, dn1, agg1);
    }
    res1<<<(N * 64 + 255) / 256, 256, 0, stream>>>(agg1, b1, h0, N);  // h0 <- h2

    // ---- conv2 (heads=1), accumulate straight into d_out ----
    gemm_wmma<64, 64, false><<<gemm_blocks, 256, 0, stream>>>(h0, w2, nullptr, h1, N);
    {
        int blocks = (N * 32 + 255) / 256;
        att_coef<1><<<blocks, 256, 0, stream>>>(h1, att_src2, att_dst2, a1s, a1d, N);
    }
    hipMemsetAsync(m1,  0, (size_t)N * sizeof(unsigned), stream);
    hipMemsetAsync(dn1, 0, (size_t)N * sizeof(float),    stream);
    hipMemsetAsync(out, 0, (size_t)out_size * sizeof(float), stream);
    edge_logit<1><<<eb, 256, 0, stream>>>(src, dst, E, Etot, a1s, a1d, e1, m1);
    edge_exp<1><<<eb, 256, 0, stream>>>(src, dst, E, Etot, m1, e1, dn1);
    {
        long long total  = (long long)Etot * 64;
        int       blocks = (int)((total + 255) / 256);
        edge_msg<1><<<blocks, 256, 0, stream>>>(src, dst, E, Etot, h1, e1, dn1, out);
    }
    bias_add<<<(N * 64 + 255) / 256, 256, 0, stream>>>(out, b2, N);
}
